// TTMultiheadAttention_65833258713656
// MI455X (gfx1250) — matrix operand order repaired and checked by reading, not compiled, not run on hardware
//
#include <hip/hip_runtime.h>
#include <hip/hip_bf16.h>

// ---------------------------------------------------------------------------
// TT Multi-head attention for MI455X (gfx1250, wave32, WMMA).
//   B=4, S=2048, E=1024, H=16, D=64.
// All GEMMs run on v_wmma_f32_16x16x32_bf16 with f32 accumulation.
// Flash attention computes S^T = K*Q^T so the softmax reduction runs along
// the in-lane register axis (1 cross-lane shuffle per 32-key block), and
// O^T = V*P so every WMMA operand is a contiguous 16/32B load.
// Inner loops are software-pipelined: next block's fragments load while the
// current block computes; global_prefetch covers the block after that.
// Workspace (bf16): Wq,Wk,Wv,Wo (1M each) | Q (8M) | K (8M) |
//   V^T [B,H,D,S] (8M) | attn-out (8M)  => 72 MB total.
// ---------------------------------------------------------------------------

typedef __attribute__((ext_vector_type(16))) __bf16 v16bf;
typedef __attribute__((ext_vector_type(8)))  float  v8f;

constexpr int BB = 4;
constexpr int SS = 2048;
constexpr int EE = 1024;
constexpr int HH = 16;
constexpr int DD = 64;
constexpr int MM = BB * SS;   // 8192 rows

union BF8 { uint4 u4; __bf16 h[8]; };

__device__ __forceinline__ __bf16 f2bf(float f) {
  unsigned u = __float_as_uint(f);
  u += 0x7FFFu + ((u >> 16) & 1u);          // round-to-nearest-even
  unsigned short s = (unsigned short)(u >> 16);
  return __builtin_bit_cast(__bf16, s);
}

__device__ __forceinline__ v16bf combine8(BF8 lo, BF8 hi) {
  v16bf r;
#pragma unroll
  for (int i = 0; i < 8; ++i) { r[i] = lo.h[i]; r[i + 8] = hi.h[i]; }
  return r;
}

__device__ __forceinline__ void cvt4(__bf16* d, float4 f) {
  d[0] = f2bf(f.x); d[1] = f2bf(f.y); d[2] = f2bf(f.z); d[3] = f2bf(f.w);
}

__device__ __forceinline__ v8f wmma_bf16(v16bf a, v16bf b, v8f c) {
  return __builtin_amdgcn_wmma_f32_16x16x32_bf16(
      /*neg_a=*/false, a, /*neg_b=*/false, b,
      /*c_mod=*/(short)0, c, /*reuse_a=*/false, /*reuse_b=*/false);
}

// B-style fragment (32x16): lane = column, K halves at +0..7 and +16..23.
__device__ __forceinline__ v16bf load_bfrag(const __bf16* p) {
  BF8 lo, hi;
  lo.u4 = *(const uint4*)(p);
  hi.u4 = *(const uint4*)(p + 16);
  return combine8(lo, hi);
}

// A-style fragment (16x32): lane = row, K halves at +0..7 and +16..23.
__device__ __forceinline__ v16bf load_afrag(const __bf16* p) {
  BF8 lo, hi;
  lo.u4 = *(const uint4*)(p);
  hi.u4 = *(const uint4*)(p + 16);
  return combine8(lo, hi);
}

// ---------------------------------------------------------------------------
// f32 -> bf16 conversion (weights)
// ---------------------------------------------------------------------------
__global__ void cvt_f32_bf16_kernel(const float* __restrict__ in,
                                    __bf16* __restrict__ out, int n) {
  int i = blockIdx.x * blockDim.x + threadIdx.x;
  int stride = gridDim.x * blockDim.x;
  for (; i < n; i += stride) out[i] = f2bf(in[i]);
}

// ---------------------------------------------------------------------------
// C[M,N] = A[M,K] * W[N,K]^T   (both K-contiguous; WMMA bf16, f32 accum)
// AKIND: 0 = A is f32 (convert on load), 1 = A is bf16
// OKIND: 0 = store bf16 row-major [M,N]
//        1 = store bf16 transposed into V^T layout [B,H,D,S]
//        2 = store f32 row-major [M,N]
// Block: 256 threads (8 waves), tile 128(M) x 64(N); wave tile 32x32.
// k-loop is unrolled x2 with ping-pong fragment sets (loads run one step
// ahead of the WMMAs that consume them).
// ---------------------------------------------------------------------------
template <int AKIND, int OKIND>
__global__ __launch_bounds__(256)
void gemm_bt_kernel(const void* __restrict__ Av,
                    const __bf16* __restrict__ Bw,
                    void* __restrict__ Out,
                    int Mtot, int Ntot, int Ktot) {
  const int lane = threadIdx.x & 31;
  const int wid  = threadIdx.x >> 5;     // 0..7
  const int wm   = wid >> 1;             // 0..3
  const int wn   = wid & 1;              // 0..1
  const int l16  = lane & 15;
  const int lh   = lane >> 4;            // 0/1

  const int rowBase = blockIdx.x * 128 + wm * 32;
  const int colBase = blockIdx.y * 64  + wn * 32;

  // lane base pointers (K-offset added per step)
  const size_t aRow0 = (size_t)(rowBase + l16) * Ktot;
  const size_t aRow1 = (size_t)(rowBase + 16 + l16) * Ktot;
  const size_t bCol0 = (size_t)(colBase + l16) * Ktot;
  const size_t bCol1 = (size_t)(colBase + 16 + l16) * Ktot;

  auto loadA = [&](v16bf (&a)[2], int k0) {
    const int koff = k0 + lh * 8;
#pragma unroll
    for (int i = 0; i < 2; ++i) {
      const size_t base = (i ? aRow1 : aRow0) + koff;
      if constexpr (AKIND == 0) {
        const float* ap = (const float*)Av + base;
        __bf16 t[16];
        cvt4(t + 0,  *(const float4*)(ap + 0));
        cvt4(t + 4,  *(const float4*)(ap + 4));
        cvt4(t + 8,  *(const float4*)(ap + 16));
        cvt4(t + 12, *(const float4*)(ap + 20));
        v16bf v;
#pragma unroll
        for (int e = 0; e < 16; ++e) v[e] = t[e];
        a[i] = v;
      } else {
        a[i] = load_afrag((const __bf16*)Av + base);
      }
    }
  };
  auto loadB = [&](v16bf (&b)[2], int k0) {
    const int koff = k0 + lh * 8;
    b[0] = load_bfrag(Bw + bCol0 + koff);
    b[1] = load_bfrag(Bw + bCol1 + koff);
  };

  v8f c[2][2] = {};
  v16bf aP[2], bP[2], aQ[2], bQ[2];
  loadA(aP, 0);
  loadB(bP, 0);

  for (int k0 = 0; k0 < Ktot; k0 += 64) {
    // loads for k0+32 in flight while computing on k0
    loadA(aQ, k0 + 32);
    loadB(bQ, k0 + 32);
    if (k0 + 64 < Ktot) {
      if constexpr (AKIND == 0)
        __builtin_prefetch((const float*)Av + aRow0 + k0 + 64, 0, 1);
      else
        __builtin_prefetch((const __bf16*)Av + aRow0 + k0 + 64, 0, 1);
      __builtin_prefetch(Bw + bCol0 + k0 + 64, 0, 1);
    }
#pragma unroll
    for (int i = 0; i < 2; ++i)
#pragma unroll
      for (int j = 0; j < 2; ++j)
        c[i][j] = wmma_bf16(aP[i], bP[j], c[i][j]);

    // loads for k0+64 in flight while computing on k0+32
    if (k0 + 64 < Ktot) {
      loadA(aP, k0 + 64);
      loadB(bP, k0 + 64);
    }
#pragma unroll
    for (int i = 0; i < 2; ++i)
#pragma unroll
      for (int j = 0; j < 2; ++j)
        c[i][j] = wmma_bf16(aQ[i], bQ[j], c[i][j]);
  }

  // ---- epilogue ----
#pragma unroll
  for (int i = 0; i < 2; ++i) {
#pragma unroll
    for (int j = 0; j < 2; ++j) {
#pragma unroll
      for (int r = 0; r < 8; ++r) {
        const int row = rowBase + i * 16 + r + 8 * lh;
        const int col = colBase + j * 16 + l16;
        const float v = c[i][j][r];
        if constexpr (OKIND == 0) {
          ((__bf16*)Out)[(size_t)row * Ntot + col] = f2bf(v);
        } else if constexpr (OKIND == 1) {
          const int bI = row >> 11;          // /S
          const int s  = row & (SS - 1);
          const int h  = col >> 6;           // /D
          const int d  = col & (DD - 1);
          ((__bf16*)Out)[((size_t)((bI * HH + h) * DD + d)) * SS + s] = f2bf(v);
        } else {
          ((float*)Out)[(size_t)row * Ntot + col] = v;
        }
      }
    }
  }
}

// ---------------------------------------------------------------------------
// Flash attention, transposed-score formulation, software-pipelined.
// Block = one (b,h) pair + 64 query columns; 4 waves, 16 queries per wave.
// Per 32-key block:
//   S^T[32k x 16q] = K-block * Q^T      (4 wmma; K-frags preloaded last iter)
//   V-frags + next K-frags issued       (latency hidden under softmax VALU)
//   softmax over k : in-lane reduction + ONE shfl_xor(16)
//   P^T -> LDS as P[q][k] (bf16)        (B-frag becomes contiguous ds_load)
//   O^T[64d x 16q] += V-block * P       (4 wmma)
// ---------------------------------------------------------------------------
__global__ __launch_bounds__(128)
void flash_attn_kernel(const __bf16* __restrict__ qg,
                       const __bf16* __restrict__ kg,
                       const __bf16* __restrict__ vT,
                       __bf16* __restrict__ og) {
  __shared__ __attribute__((aligned(16))) __bf16 plds[4][16][32];  // [wave][q][k]

  const int lane = threadIdx.x & 31;
  const int wid  = threadIdx.x >> 5;   // 0..3
  const int l16  = lane & 15;
  const int lh   = lane >> 4;

  const int qblk = blockIdx.x & 31;        // S/64 query blocks
  const int bh   = blockIdx.x >> 5;        // 0..63
  const int b    = bh >> 4;
  const int h    = bh & (HH - 1);
  const int q0   = qblk * 64 + wid * 16;   // first query of this wave

  const float sm_scale = 0.125f;           // D^-0.5

  // lane base pointers
  const __bf16* kRow = kg + (size_t)(b * SS + l16) * EE + h * DD + lh * 8;
  const __bf16* vRow0 = vT + (size_t)(bh * DD + l16) * SS + lh * 8;

  // ---- Q B-fragments: lane = query column, d-chunk = t*32 + lh*8 (+16) ----
  v16bf qb[2];
#pragma unroll
  for (int t = 0; t < 2; ++t)
    qb[t] = load_bfrag(qg + (size_t)(b * SS + q0 + l16) * EE
                          + h * DD + t * 32 + lh * 8);

  auto loadK = [&](v16bf (&kf)[2][2], int kb) {
#pragma unroll
    for (int mt = 0; mt < 2; ++mt)
#pragma unroll
      for (int t = 0; t < 2; ++t)
        kf[mt][t] = load_afrag(kRow + (size_t)(kb + mt * 16) * EE + t * 32);
  };

  float mrun = -1e30f, lrun = 0.0f;        // per-lane scalars (one query)
  v8f o[4] = {};                           // O^T: rows d = t4*16 + r + 8*lh

  v16bf kcur[2][2];
  loadK(kcur, 0);

  for (int kb = 0; kb < SS; kb += 32) {
    // ---- S^T: two 16-key row tiles, K-dim = D = 64 (2 wmma each) ----
    v8f st[2];
#pragma unroll
    for (int mt = 0; mt < 2; ++mt) {
      v8f acc = {};
      acc = wmma_bf16(kcur[mt][0], qb[0], acc);
      acc = wmma_bf16(kcur[mt][1], qb[1], acc);
      st[mt] = acc;
    }

    // ---- V-frags for this block + K-frags for next block (in flight) ----
    v16bf va[4];
#pragma unroll
    for (int t4 = 0; t4 < 4; ++t4)
      va[t4] = load_afrag(vRow0 + (size_t)(t4 * 16) * SS + kb);
    if (kb + 32 < SS) loadK(kcur, kb + 32);
    if (kb + 64 < SS) {
      __builtin_prefetch(kRow + (size_t)(kb + 64) * EE, 0, 1);
      __builtin_prefetch(kRow + (size_t)(kb + 80) * EE, 0, 1);
      __builtin_prefetch(vRow0 + kb + 64, 0, 1);
      __builtin_prefetch(vRow0 + (size_t)(32) * SS + kb + 64, 0, 1);
    }

    // ---- softmax over the 32 keys: in-lane + one half-wave merge ----
    float sc[16];
#pragma unroll
    for (int mt = 0; mt < 2; ++mt)
#pragma unroll
      for (int r = 0; r < 8; ++r)
        sc[mt * 8 + r] = st[mt][r] * sm_scale;

    float mb = sc[0];
#pragma unroll
    for (int e = 1; e < 16; ++e) mb = fmaxf(mb, sc[e]);
    mb = fmaxf(mb, __shfl_xor(mb, 16, 32));          // merge lane halves

    const float mnew = fmaxf(mrun, mb);
    const float corr = __expf(mrun - mnew);
    float p[16], rs = 0.0f;
#pragma unroll
    for (int e = 0; e < 16; ++e) { p[e] = __expf(sc[e] - mnew); rs += p[e]; }
    rs += __shfl_xor(rs, 16, 32);                    // merge lane halves
    lrun = lrun * corr + rs;
    mrun = mnew;
#pragma unroll
    for (int t4 = 0; t4 < 4; ++t4)
#pragma unroll
      for (int r = 0; r < 8; ++r) o[t4][r] *= corr;  // per-lane broadcast

    // ---- P^T (C-layout) -> LDS tile P[q][k] ----
#pragma unroll
    for (int mt = 0; mt < 2; ++mt)
#pragma unroll
      for (int r = 0; r < 8; ++r)
        plds[wid][l16][mt * 16 + r + 8 * lh] = f2bf(p[mt * 8 + r]);
    asm volatile("s_wait_dscnt 0" ::: "memory");

    // ---- P B-fragment: lane = query column, k at lh*8+0..7 and +16 ----
    const v16bf pb = load_bfrag(&plds[wid][l16][lh * 8]);

    // ---- O^T += V-block * P ----
#pragma unroll
    for (int t4 = 0; t4 < 4; ++t4)
      o[t4] = wmma_bf16(va[t4], pb, o[t4]);
    asm volatile("s_wait_dscnt 0" ::: "memory");     // P consumed before reuse
  }

  // ---- normalize + store bf16 [B,S,E]; lane writes its query's channels ----
  const float inv = 1.0f / lrun;
  const size_t orow = (size_t)(b * SS + q0 + l16) * EE + h * DD;
#pragma unroll
  for (int t4 = 0; t4 < 4; ++t4)
#pragma unroll
    for (int r = 0; r < 8; ++r)
      og[orow + t4 * 16 + r + 8 * lh] = f2bf(o[t4][r] * inv);
}

// ---------------------------------------------------------------------------
// Host launcher
// ---------------------------------------------------------------------------
extern "C" void kernel_launch(void* const* d_in, const int* in_sizes, int n_in,
                              void* d_out, int out_size, void* d_ws, size_t ws_size,
                              hipStream_t stream) {
  (void)in_sizes; (void)n_in; (void)out_size; (void)ws_size;

  const float* query = (const float*)d_in[0];
  const float* keyi  = (const float*)d_in[1];
  const float* value = (const float*)d_in[2];
  const float* Wq    = (const float*)d_in[3];
  const float* Wk    = (const float*)d_in[4];
  const float* Wv    = (const float*)d_in[5];
  const float* Wo    = (const float*)d_in[6];
  float* out = (float*)d_out;

  // workspace carve-up (bf16 elements)
  __bf16* wq_bf   = (__bf16*)d_ws;
  __bf16* wk_bf   = wq_bf  + (size_t)EE * EE;
  __bf16* wv_bf   = wk_bf  + (size_t)EE * EE;
  __bf16* wo_bf   = wv_bf  + (size_t)EE * EE;
  __bf16* q_bf    = wo_bf  + (size_t)EE * EE;      // [B,S,E]
  __bf16* k_bf    = q_bf   + (size_t)MM * EE;      // [B,S,E]
  __bf16* vT_bf   = k_bf   + (size_t)MM * EE;      // [B,H,D,S]
  __bf16* attn_bf = vT_bf  + (size_t)MM * EE;      // [B,S,E]

  // 1) weights -> bf16
  cvt_f32_bf16_kernel<<<1024, 256, 0, stream>>>(Wq, wq_bf, EE * EE);
  cvt_f32_bf16_kernel<<<1024, 256, 0, stream>>>(Wk, wk_bf, EE * EE);
  cvt_f32_bf16_kernel<<<1024, 256, 0, stream>>>(Wv, wv_bf, EE * EE);
  cvt_f32_bf16_kernel<<<1024, 256, 0, stream>>>(Wo, wo_bf, EE * EE);

  const dim3 gdim(MM / 128, EE / 64);
  // 2) projections (A = f32 input, converted on load)
  gemm_bt_kernel<0, 0><<<gdim, 256, 0, stream>>>(query, wq_bf, q_bf,  MM, EE, EE);
  gemm_bt_kernel<0, 0><<<gdim, 256, 0, stream>>>(keyi,  wk_bf, k_bf,  MM, EE, EE);
  gemm_bt_kernel<0, 1><<<gdim, 256, 0, stream>>>(value, wv_bf, vT_bf, MM, EE, EE);

  // 3) fused flash attention (scores never touch memory)
  flash_attn_kernel<<<BB * HH * (SS / 64), 128, 0, stream>>>(q_bf, k_bf, vT_bf, attn_bf);

  // 4) output projection -> f32
  gemm_bt_kernel<1, 2><<<gdim, 256, 0, stream>>>(attn_bf, wo_bf, out, MM, EE, EE);
}
